// STAAttention_82222853914754
// MI455X (gfx1250) — matrix operand order repaired and checked by reading, not compiled, not run on hardware
//
#include <hip/hip_runtime.h>

typedef __attribute__((ext_vector_type(16))) _Float16 v16h;
typedef __attribute__((ext_vector_type(8)))  _Float16 v8h;
typedef __attribute__((ext_vector_type(8)))  float    v8f;
typedef __attribute__((ext_vector_type(4)))  int      v4i;

#define AS1 __attribute__((address_space(1)))
#define AS3 __attribute__((address_space(3)))

#define DDIM   128
#define SEQ    10688
#define IMGLEN 10368
#define NBLK   28
#define SCALE_F 0.08838834764831845f   // 1/sqrt(128)

#if __has_builtin(__builtin_amdgcn_global_load_async_to_lds_b128)
#define HAVE_ASYNC 1
#else
#define HAVE_ASYNC 0
#endif

#if __has_builtin(__builtin_amdgcn_s_wait_asynccnt)
#define WAIT_ASYNC0() __builtin_amdgcn_s_wait_asynccnt(0)
#else
#define WAIT_ASYNC0() asm volatile("s_wait_asynccnt 0" ::: "memory")
#endif

// 16-byte LDS <- global copy; async DMA path on gfx1250, sync fallback otherwise
__device__ __forceinline__ void cp16_g2l(_Float16* lds_dst, const _Float16* gsrc) {
#if HAVE_ASYNC
    __builtin_amdgcn_global_load_async_to_lds_b128(
        (AS1 v4i*)gsrc, (AS3 v4i*)lds_dst, 0, 0);
#else
    *(v8h*)lds_dst = *(const v8h*)gsrc;
#endif
}

// ---- 16-lane butterfly reductions in pure VALU via DPP16 (no LDS round trips) ----
#if __has_builtin(__builtin_amdgcn_mov_dpp)
#define HAVE_DPP 1
template <int CTRL>
__device__ __forceinline__ float dppf(float x) {
    return __int_as_float(__builtin_amdgcn_mov_dpp(__float_as_int(x), CTRL, 0xF, 0xF, true));
}
__device__ __forceinline__ float red_max16(float x) {
    x = fmaxf(x, dppf<0xB1>(x));   // quad_perm(1,0,3,2)  : xor 1
    x = fmaxf(x, dppf<0x4E>(x));   // quad_perm(2,3,0,1)  : xor 2
    x = fmaxf(x, dppf<0x141>(x));  // row_half_mirror     : xor 7
    x = fmaxf(x, dppf<0x140>(x));  // row_mirror          : xor 15
    return x;
}
__device__ __forceinline__ float red_sum16(float x) {
    x += dppf<0xB1>(x);
    x += dppf<0x4E>(x);
    x += dppf<0x141>(x);
    x += dppf<0x140>(x);
    return x;
}
#else
#define HAVE_DPP 0
__device__ __forceinline__ float red_max16(float x) {
    for (int m = 8; m >= 1; m >>= 1) x = fmaxf(x, __shfl_xor(x, m, 32));
    return x;
}
__device__ __forceinline__ float red_sum16(float x) {
    for (int m = 8; m >= 1; m >>= 1) x += __shfl_xor(x, m, 32);
    return x;
}
#endif

// tiled (block n, pos p) -> original token index; -1 = zero pad
__device__ __forceinline__ int tok_orig(int n, int p) {
    if (n < 27) {
        int nt = n / 9, nh = (n % 9) / 3, nw = n % 3;
        int tt = p >> 6, th = (p >> 3) & 7, tw = p & 7;
        return ((nt * 6 + tt) * 24 + (nh * 8 + th)) * 24 + (nw * 8 + tw);
    }
    return (p < 320) ? (IMGLEN + p) : -1;
}

// A-matrix 16x32 f16 fragment from LDS tile [16][stride] (row-major).
__device__ __forceinline__ v16h fragA(const _Float16* base, int stride, int lane) {
    int row = lane & 15;
    int off = (lane & 16) ? 8 : 0;
    const _Float16* p = base + row * stride + off;
    v8h lo = *(const v8h*)(p);
    v8h hi = *(const v8h*)(p + 16);
    v16h r;
#pragma unroll
    for (int i = 0; i < 8; ++i) { r[i] = lo[i]; r[8 + i] = hi[i]; }
    return r;
}

// B-matrix 32x16 f16 fragment from LDS tile where base row = column N.
__device__ __forceinline__ v16h fragB(const _Float16* base, int stride, int lane) {
    int row = lane & 15;
    int off = (lane & 16) ? 8 : 0;
    const _Float16* p = base + row * stride + off;
    v8h lo = *(const v8h*)(p);
    v8h hi = *(const v8h*)(p + 16);
    v16h r;
#pragma unroll
    for (int i = 0; i < 8; ++i) { r[i] = lo[i]; r[8 + i] = hi[i]; }
    return r;
}

__device__ __forceinline__ v8h cvt8(float4 a, float4 b) {
    v8h h;
    h[0] = (_Float16)a.x; h[1] = (_Float16)a.y; h[2] = (_Float16)a.z; h[3] = (_Float16)a.w;
    h[4] = (_Float16)b.x; h[5] = (_Float16)b.y; h[6] = (_Float16)b.z; h[7] = (_Float16)b.w;
    return h;
}

// ---------------------------------------------------------------------------
// Prep: one-shot fp32 -> f16 conversion into tiled workspace.
//   Qf, Kf : [h][28*384][128] f16 (tiled token order, zero-padded)
//   Vt     : [h][28][128][384] f16 (per-block transposed)
// ---------------------------------------------------------------------------
__global__ __launch_bounds__(128)
void sta_prep_kernel(const float* __restrict__ q, const float* __restrict__ k,
                     const float* __restrict__ v,
                     _Float16* __restrict__ Qf, _Float16* __restrict__ Kf,
                     _Float16* __restrict__ Vt) {
    const int w = blockIdx.x;
    const int h   = w / (NBLK * 12);
    const int rem = w % (NBLK * 12);
    const int n   = rem / 12;
    const int pc  = rem % 12;
    const int tid = threadIdx.x;
    const int p   = pc * 32 + (tid >> 2);
    const int c0  = (tid & 3) * 32;
    const int tok = tok_orig(n, p);

    const size_t goff = (size_t)h * SEQ * DDIM + ((tok < 0) ? 0 : (size_t)tok * DDIM) + c0;
    const float4* qs = (const float4*)(q + goff);
    const float4* ks = (const float4*)(k + goff);
    const float4* vs = (const float4*)(v + goff);

    const size_t row = (size_t)(h * NBLK + n) * 384 + p;
    _Float16* qd = Qf + row * DDIM + c0;
    _Float16* kd = Kf + row * DDIM + c0;

#pragma unroll
    for (int i = 0; i < 4; ++i) {
        float4 a = qs[2 * i], b = qs[2 * i + 1];
        float4 c = ks[2 * i], d = ks[2 * i + 1];
        if (tok < 0) { a = make_float4(0.f, 0.f, 0.f, 0.f); b = a; c = a; d = a; }
        *(v8h*)(qd + i * 8) = cvt8(a, b);
        *(v8h*)(kd + i * 8) = cvt8(c, d);
    }
#pragma unroll
    for (int i = 0; i < 8; ++i) {
        float4 a = vs[i];
        if (tok < 0) a = make_float4(0.f, 0.f, 0.f, 0.f);
        const size_t vrow = ((size_t)(h * NBLK + n) * DDIM + (c0 + i * 4));
        Vt[(vrow + 0) * 384 + p] = (_Float16)a.x;
        Vt[(vrow + 1) * 384 + p] = (_Float16)a.y;
        Vt[(vrow + 2) * 384 + p] = (_Float16)a.z;
        Vt[(vrow + 3) * 384 + p] = (_Float16)a.w;
    }
}

// ---------------------------------------------------------------------------
// Flash attention with WMMA, double-buffered async-DMA pipeline.
// 672 wgs * 128 threads (4 waves, 16 query rows/wave).
// ---------------------------------------------------------------------------
__global__ __launch_bounds__(128)
void sta_attn_kernel(const _Float16* __restrict__ Qf, const _Float16* __restrict__ Kf,
                     const _Float16* __restrict__ Vt, float* __restrict__ out) {
    __shared__ _Float16 sQ[64 * DDIM];        // 16 KB
    __shared__ _Float16 sK[2][32 * DDIM];     // 16 KB
    __shared__ _Float16 sVt[2][DDIM * 32];    // 16 KB
    __shared__ _Float16 sP[4][16 * 32];       // 4 KB

    const int w = blockIdx.x;
    const bool istext = (w >= 648);
    int h, qblk = 27, slice, qt = 0, nkv;
    if (!istext) {
        h = w / 162; int rem = w % 162;
        qblk = rem / 6; slice = rem % 6; qt = qblk / 9; nkv = 10;
    } else {
        int w2 = w - 648; h = w2 / 6; slice = w2 % 6; nkv = 28;
    }

    const int tid  = threadIdx.x;
    const int lane = tid & 31;
    const int wave = tid >> 5;
    const int TSTEPS = nkv * 12;              // 32-key tiles

    // async-stage a 32-key K/Vt tile into buffer `buf`
    auto issue_tile = [&](int t, int buf) {
        int bi = t / 12, p0 = (t % 12) * 32;
        int n = istext ? bi : ((bi < 9) ? (qt * 9 + bi) : 27);
        const _Float16* ks   = Kf + ((size_t)(h * NBLK + n) * 384 + p0) * DDIM;
        const _Float16* vsrc = Vt + ((size_t)(h * NBLK + n) * DDIM + tid) * 384 + p0;
#pragma unroll
        for (int i = 0; i < 4; ++i) {
            int c = i * 128 + tid;            // 512 chunks of 16B, contiguous 8KB
            cp16_g2l(&sK[buf][c * 8], ks + c * 8);
        }
#pragma unroll
        for (int i = 0; i < 4; ++i)           // 128 d-rows x 32 keys
            cp16_g2l(&sVt[buf][tid * 32 + i * 8], vsrc + i * 8);
    };

    // ---- prologue: stage Q slice (contiguous 16KB) + tile 0 ----
    {
        const _Float16* qsrc = Qf + ((size_t)(h * NBLK + qblk) * 384 + slice * 64) * DDIM;
#pragma unroll
        for (int i = 0; i < 8; ++i) {
            int c = i * 128 + tid;
            cp16_g2l(&sQ[c * 8], qsrc + c * 8);
        }
        issue_tile(0, 0);
        WAIT_ASYNC0();
    }
    __syncthreads();

    v16h aQ[4];
#pragma unroll
    for (int c = 0; c < 4; ++c)
        aQ[c] = fragA(&sQ[(wave * 16) * DDIM + 32 * c], DDIM, lane);

    v8f   o[8];
    float mrun[8], lrun[8];
#pragma unroll
    for (int j = 0; j < 8; ++j) { v8f z = {}; o[j] = z; }
#pragma unroll
    for (int r = 0; r < 8; ++r) { mrun[r] = -3.0e38f; lrun[r] = 0.f; }

    _Float16* sPw = sP[wave];

    for (int t = 0; t < TSTEPS; ++t) {
        const int cur = t & 1;
        if (t + 1 < TSTEPS) issue_tile(t + 1, cur ^ 1);   // DMA overlaps compute

        // S = Q (16x128) x K^T (128x32) -> two 16x16 f32 tiles
        v8f s0 = {}, s1 = {};
#pragma unroll
        for (int c = 0; c < 4; ++c) {
            v16h b0 = fragB(&sK[cur][0 * DDIM + 32 * c], DDIM, lane);
            v16h b1 = fragB(&sK[cur][16 * DDIM + 32 * c], DDIM, lane);
            s0 = __builtin_amdgcn_wmma_f32_16x16x32_f16(false, aQ[c], false, b0, (short)0, s0, false, false);
            s1 = __builtin_amdgcn_wmma_f32_16x16x32_f16(false, aQ[c], false, b1, (short)0, s1, false, false);
        }

        // online softmax (DPP butterfly reductions, no LDS)
#pragma unroll
        for (int r = 0; r < 8; ++r) {
            float a0 = s0[r] * SCALE_F, a1 = s1[r] * SCALE_F;
            float mx    = red_max16(fmaxf(a0, a1));
            float mnew  = fmaxf(mrun[r], mx);
            float alpha = __expf(mrun[r] - mnew);
            float e0 = __expf(a0 - mnew);
            float e1 = __expf(a1 - mnew);
            float rs = red_sum16(e0 + e1);
            lrun[r] = lrun[r] * alpha + rs;
            mrun[r] = mnew;
#pragma unroll
            for (int j = 0; j < 8; ++j) o[j][r] *= alpha;
            int prow = r + ((lane & 16) ? 8 : 0);
            sPw[prow * 32 + (lane & 15)]      = (_Float16)e0;
            sPw[prow * 32 + 16 + (lane & 15)] = (_Float16)e1;
        }

        // O += P (16x32) x V (32x128)
        v16h aP = fragA(sPw, 32, lane);
#pragma unroll
        for (int j = 0; j < 8; ++j) {
            v16h bV = fragB(&sVt[cur][(j * 16) * 32], 32, lane);
            o[j] = __builtin_amdgcn_wmma_f32_16x16x32_f16(false, aP, false, bV, (short)0, o[j], false, false);
        }

        WAIT_ASYNC0();        // next tile's DMA has been hidden under compute
        __syncthreads();
    }

    // ---- normalize + scatter to untiled fp32 output ----
    float* ob = out + (size_t)h * SEQ * DDIM;
#pragma unroll
    for (int r = 0; r < 8; ++r) {
        int row = wave * 16 + r + ((lane & 16) ? 8 : 0);
        int p   = slice * 64 + row;
        int tok = istext ? ((p < 320) ? (IMGLEN + p) : -1) : tok_orig(qblk, p);
        if (tok >= 0) {
            float inv = 1.0f / lrun[r];
            float* dst = ob + (size_t)tok * DDIM + (lane & 15);
#pragma unroll
            for (int j = 0; j < 8; ++j)
                dst[j * 16] = o[j][r] * inv;
        }
    }
}

extern "C" void kernel_launch(void* const* d_in, const int* in_sizes, int n_in,
                              void* d_out, int out_size, void* d_ws, size_t ws_size,
                              hipStream_t stream) {
    const float* q = (const float*)d_in[0];
    const float* k = (const float*)d_in[1];
    const float* v = (const float*)d_in[2];
    float* out = (float*)d_out;
    (void)in_sizes; (void)n_in; (void)out_size; (void)ws_size;

    const size_t HBLK = (size_t)4 * NBLK * 384 * DDIM;   // halves per tensor
    _Float16* Qf = (_Float16*)d_ws;
    _Float16* Kf = Qf + HBLK;
    _Float16* Vt = Kf + HBLK;

    hipLaunchKernelGGL(sta_prep_kernel, dim3(4 * NBLK * 12), dim3(128), 0, stream,
                       q, k, v, Qf, Kf, Vt);
    hipLaunchKernelGGL(sta_attn_kernel, dim3(672), dim3(128), 0, stream,
                       Qf, Kf, Vt, out);
}
